// Decoder_13322988552723
// MI455X (gfx1250) — matrix operand order repaired and checked by reading, not compiled, not run on hardware
//
#include <hip/hip_runtime.h>
#include <hip/hip_bf16.h>
#include <stdint.h>

// Problem constants (B=1)
#define S_LEN 2048
#define EMB   4096
#define NHEAD 32
#define HDIM  128
#define KEEP  205
#define KEEP_PAD 208

typedef __attribute__((ext_vector_type(16))) __bf16 v16bf;
typedef __attribute__((ext_vector_type(8)))  float  v8f;

// ---------- small helpers ----------
__device__ inline unsigned short f2bf(float f) {
    unsigned u = __float_as_uint(f);
    u += 0x7FFFu + ((u >> 16) & 1u);          // round-to-nearest-even
    return (unsigned short)(u >> 16);
}
__device__ inline float bf2f(unsigned short h) {
    return __uint_as_float(((unsigned)h) << 16);
}
// monotone sortable key for bf16 bits
__device__ inline unsigned bkey(unsigned short u) {
    return (u & 0x8000u) ? (unsigned)((unsigned short)~u)
                         : (unsigned)(u | 0x8000u);
}
__device__ inline unsigned wave_sum_u(unsigned v) {
#pragma unroll
    for (int o = 16; o; o >>= 1) v += __shfl_xor(v, o, 32);
    return v;
}
__device__ inline float wave_sum_f(float v) {
#pragma unroll
    for (int o = 16; o; o >>= 1) v += __shfl_xor(v, o, 32);
    return v;
}
__device__ inline float wave_max_f(float v) {
#pragma unroll
    for (int o = 16; o; o >>= 1) v = fmaxf(v, __shfl_xor(v, o, 32));
    return v;
}

// CDNA5 async global->LDS copy (ASYNCcnt-tracked, no VGPR staging).
// LDS destination address = low 32 bits of the generic pointer (ISA 10.2:
// LDS aperture maps addr[31:0] to the LDS offset).
__device__ inline void async_ld_b128(unsigned lds_off, const void* gaddr) {
    asm volatile("global_load_async_to_lds_b128 %0, %1, off"
                 :: "v"(lds_off), "v"(gaddr) : "memory");
}
__device__ inline void wait_async0() {
    asm volatile("s_wait_asynccnt 0x0" ::: "memory");
}

// ---------- kernel 1a: fp32 -> bf16 cast (row-major copy) ----------
__global__ void cast_f32_bf16(const float* __restrict__ in,
                              unsigned short* __restrict__ out, int n) {
    int i = blockIdx.x * 256 + threadIdx.x;
    if (i < n) out[i] = f2bf(in[i]);
}

// ---------- kernel 1b: fp32 [K][N] -> bf16 transposed [N][K] ----------
__global__ __launch_bounds__(256) void transpose_cast_bf16(
    const float* __restrict__ in, unsigned short* __restrict__ out) {
    __shared__ float tile[32][33];
    const int tx = threadIdx.x & 31, ty = threadIdx.x >> 5;  // 32x8
    const int n0 = blockIdx.x * 32, k0 = blockIdx.y * 32;
#pragma unroll
    for (int i = ty; i < 32; i += 8)
        tile[i][tx] = in[(size_t)(k0 + i) * EMB + n0 + tx];
    __syncthreads();
#pragma unroll
    for (int i = ty; i < 32; i += 8)
        out[(size_t)(n0 + i) * EMB + k0 + tx] = f2bf(tile[tx][i]);
}

// ---------- kernel 2: tiled bf16 WMMA GEMM, async double-buffered ----------
// C[M,N] = A[M,K] * B^T, A row-major [M][K], Bt row-major [N][K], fp32 accum.
// mode 0: out fp32 head-major [h][s][d] (n = h*128+d)
// mode 1: out fp32 row-major [m][n]
// mode 2: out bf16 head-major [h][s][d]
#define BM 128
#define BN 128
#define BK 32
#define LDSS (BK + 8)

__global__ __launch_bounds__(256) void gemm_bf16_wmma(
    const unsigned short* __restrict__ A,
    const unsigned short* __restrict__ Bt,
    void* __restrict__ Out, int M, int K, int N, int mode) {
    __shared__ unsigned short sA[2][BM][LDSS];
    __shared__ unsigned short sB[2][BN][LDSS];

    const int row0 = blockIdx.x * BM;
    const int col0 = blockIdx.y * BN;
    const int t = threadIdx.x;
    const int lane = t & 31, wave = t >> 5;
    const int wm = (wave & 3) * 32;        // wave sub-tile: 32(M) x 64(N)
    const int wn = (wave >> 2) * 64;

    v8f acc[2][4];
#pragma unroll
    for (int i = 0; i < 2; ++i)
#pragma unroll
        for (int j = 0; j < 4; ++j) acc[i][j] = (v8f){};

    // async tile loaders: 128x32 bf16 per operand = 512 b128 transfers
    auto issue = [&](int buf, int k0) {
#pragma unroll
        for (int r = 0; r < 2; ++r) {
            int id = t + r * 256;
            int mm = id >> 2, kq = (id & 3) * 8;
            async_ld_b128((unsigned)(size_t)&sA[buf][mm][kq],
                          A + (size_t)(row0 + mm) * K + k0 + kq);
            async_ld_b128((unsigned)(size_t)&sB[buf][mm][kq],
                          Bt + (size_t)(col0 + mm) * K + k0 + kq);
        }
    };

    issue(0, 0);
    const int nIter = K / BK;
    for (int it = 0; it < nIter; ++it) {
        const int buf = it & 1;
        wait_async0();           // my async writes into sX[buf] done
        __syncthreads();         // everyone's done -> tile valid, prev reads done
        if (it + 1 < nIter) issue(buf ^ 1, (it + 1) * BK);

        // fragments per ISA 16-bit A(16x32)/B(32x16) lane packings
        union F { v16bf v; unsigned short u[16]; } a[2], b[4];
        const int l16 = lane & 15, half = lane >> 4;
#pragma unroll
        for (int i = 0; i < 2; ++i) {
            const unsigned short* src = &sA[buf][wm + i * 16 + l16][0];
#pragma unroll
            for (int vv = 0; vv < 8; ++vv) {
                int ka = (vv < 4) ? (half * 8 + 2 * vv)
                                  : (16 + half * 8 + 2 * (vv - 4));
                unsigned p = *reinterpret_cast<const unsigned*>(src + ka);
                a[i].u[2 * vv] = (unsigned short)p;
                a[i].u[2 * vv + 1] = (unsigned short)(p >> 16);
            }
        }
#pragma unroll
        for (int j = 0; j < 4; ++j) {
            const unsigned short* src = &sB[buf][wn + j * 16 + l16][0];
#pragma unroll
            for (int vv = 0; vv < 8; ++vv) {
                int kb = (vv < 4) ? (half * 8 + 2 * vv)
                                  : (16 + half * 8 + 2 * (vv - 4));
                unsigned p = *reinterpret_cast<const unsigned*>(src + kb);
                b[j].u[2 * vv] = (unsigned short)p;
                b[j].u[2 * vv + 1] = (unsigned short)(p >> 16);
            }
        }
#pragma unroll
        for (int i = 0; i < 2; ++i)
#pragma unroll
            for (int j = 0; j < 4; ++j)
                acc[i][j] = __builtin_amdgcn_wmma_f32_16x16x32_bf16(
                    false, a[i].v, false, b[j].v, (short)0, acc[i][j], false, false);
    }

    // C/D layout: lanes 0-15 -> M=r, lanes 16-31 -> M=r+8; N = lane%16
    const int n = lane & 15;
    const int mbase = (lane >> 4) ? 8 : 0;
#pragma unroll
    for (int i = 0; i < 2; ++i) {
#pragma unroll
        for (int j = 0; j < 4; ++j) {
#pragma unroll
            for (int r = 0; r < 8; ++r) {
                int gm = row0 + wm + i * 16 + mbase + r;
                int gn = col0 + wn + j * 16 + n;
                float f = acc[i][j][r];
                if (mode == 1) {
                    ((float*)Out)[(size_t)gm * N + gn] = f;
                } else if (mode == 0) {
                    ((float*)Out)[((size_t)(gn >> 7) * S_LEN + gm) * HDIM + (gn & 127)] = f;
                } else {
                    ((unsigned short*)Out)
                        [((size_t)(gn >> 7) * S_LEN + gm) * HDIM + (gn & 127)] = f2bf(f);
                }
            }
        }
    }
}

// ---------- kernel 3: RoPE (reads pre-RoPE bf16 q/k, writes fp32 roped) ----------
__global__ __launch_bounds__(128) void rope_kernel(
    const unsigned short* __restrict__ qb, const unsigned short* __restrict__ kb,
    float* __restrict__ qr, float* __restrict__ kr) {
    const int row = blockIdx.x;            // h*S + s
    const int d = threadIdx.x;             // 0..127
    const int s = row & (S_LEN - 1);
    const int j = (d < 64) ? d : d - 64;
    const float freq = expf(-(float)(2 * j) * (1.0f / 128.0f) * 9.210340371976184f);
    const float ang = (float)s * freq;
    float c = cosf(ang), sn = sinf(ang);
    const size_t base = (size_t)row * HDIM;
    float q  = bf2f(qb[base + d]);
    float k  = bf2f(kb[base + d]);
    float qh = (d < 64) ? -bf2f(qb[base + d + 64]) : bf2f(qb[base + d - 64]);
    float kh = (d < 64) ? -bf2f(kb[base + d + 64]) : bf2f(kb[base + d - 64]);
    qr[base + d] = q * c + qh * sn;
    kr[base + d] = k * c + kh * sn;
}

// ---------- kernel 4: draft scores (WMMA) + exact top-205 per row ----------
__global__ __launch_bounds__(256) void draft_topk_kernel(
    const unsigned short* __restrict__ qb, const unsigned short* __restrict__ kb,
    int* __restrict__ topk, int* __restrict__ cnt) {
    extern __shared__ unsigned short sm[];
    unsigned short* sQ = sm;                 // 16 x 128 bf16 (q rows, pre-RoPE)
    unsigned short* sD = sm + 16 * HDIM;     // 16 x 2048 bf16 draft scores

    const int h = blockIdx.y;
    const int q0 = blockIdx.x * 16;
    const int t = threadIdx.x, lane = t & 31, wave = t >> 5;

    {   // stage the 16 query rows (one b128 per thread)
        int r = t >> 4, dq = (t & 15) * 8;
        *reinterpret_cast<uint4*>(&sQ[r * HDIM + dq]) =
            *reinterpret_cast<const uint4*>(qb + ((size_t)h * S_LEN + q0 + r) * HDIM + dq);
    }
    __syncthreads();

    const int numTiles = blockIdx.x + 1;     // causal: kv tiles 0 .. blockIdx.x
    for (int tile = wave; tile < numTiles; tile += 8) {
        v8f acc = {};
        const int m = lane & 15, half = lane >> 4;
        const unsigned short* krow = kb + ((size_t)h * S_LEN + tile * 16 + m) * HDIM;
#pragma unroll
        for (int ks = 0; ks < 4; ++ks) {     // contraction over D=128 in 4x32 steps
            union { v16bf v; unsigned short u[16]; } af, bfr;
#pragma unroll
            for (int vv = 0; vv < 8; ++vv) {
                int ka = ks * 32 + ((vv < 4) ? (half * 8 + 2 * vv)
                                             : (16 + half * 8 + 2 * (vv - 4)));
                unsigned pr = *reinterpret_cast<const unsigned*>(&sQ[m * HDIM + ka]);
                af.u[2 * vv] = (unsigned short)pr;
                af.u[2 * vv + 1] = (unsigned short)(pr >> 16);
                int kd = ks * 32 + ((vv < 4) ? (half * 8 + 2 * vv)
                                             : (16 + half * 8 + 2 * (vv - 4)));   // B: K=d same table order as A
                unsigned pb = *reinterpret_cast<const unsigned*>(krow + kd);
                bfr.u[2 * vv] = (unsigned short)pb;
                bfr.u[2 * vv + 1] = (unsigned short)(pb >> 16);
            }
            acc = __builtin_amdgcn_wmma_f32_16x16x32_bf16(false, af.v, false, bfr.v,
                                                          (short)0, acc, false, false);
        }
        const int nc = lane & 15;
        const int kvc = tile * 16 + nc;
        const int mb = (lane >> 4) ? 8 : 0;
#pragma unroll
        for (int r = 0; r < 8; ++r) {
            int mm = mb + r;
            float f = acc[r];
            if (kvc > q0 + mm) f = -3.0e38f;
            sD[mm * S_LEN + kvc] = f2bf(f);
        }
    }
    __syncthreads();

    // selection: wave w handles rows w and w+8
    for (int rr = 0; rr < 2; ++rr) {
        const int r = wave + rr * 8;
        const int q = q0 + r;
        const int ncand = q + 1;
        int* outIdx = topk + ((size_t)h * S_LEN + q) * KEEP_PAD;
        const unsigned short* row = sD + r * S_LEN;

        if (ncand <= KEEP) {
            for (int j = lane; j < ncand; j += 32) outIdx[j] = j;
            if (lane == 0) cnt[h * S_LEN + q] = ncand;
            continue;
        }
        unsigned T = 0;   // largest T with count(key >= T) >= KEEP
        for (int bit = 15; bit >= 0; --bit) {
            unsigned candT = T | (1u << bit);
            unsigned c = 0;
            for (int j = lane; j < ncand; j += 32) c += (bkey(row[j]) >= candT);
            if (wave_sum_u(c) >= KEEP) T = candT;
        }
        unsigned cntGT = 0;
        {
            unsigned c = 0;
            for (int j = lane; j < ncand; j += 32) c += (bkey(row[j]) > T);
            cntGT = wave_sum_u(c);
        }
        const int tieNeed = KEEP - (int)cntGT;
        const unsigned lmlt = (1u << lane) - 1u;
        unsigned base = 0;
        for (int j0 = 0; j0 < ncand; j0 += 32) {   // strictly greater
            int j = j0 + lane;
            bool pred = (j < ncand) && (bkey(row[j]) > T);
            unsigned msk = (unsigned)__ballot(pred);
            if (pred) outIdx[base + __popc(msk & lmlt)] = j;
            base += __popc(msk);
        }
        unsigned tb = 0;
        for (int j0 = 0; j0 < ncand && (int)tb < tieNeed; j0 += 32) {  // ties
            int j = j0 + lane;
            bool pred = (j < ncand) && (bkey(row[j]) == T);
            unsigned msk = (unsigned)__ballot(pred);
            int slot = (int)(tb + __popc(msk & lmlt));
            if (pred && slot < tieNeed) outIdx[cntGT + slot] = j;
            tb += __popc(msk);
        }
        if (lane == 0) cnt[h * S_LEN + q] = KEEP;
    }
}

// ---------- kernel 5: sparse attention over kept keys ----------
__global__ __launch_bounds__(256) void sparse_attn_kernel(
    const float* __restrict__ qr, const float* __restrict__ kr,
    const float* __restrict__ v, const int* __restrict__ topk,
    const int* __restrict__ cnt, unsigned short* __restrict__ outbf) {
    __shared__ float sq[8][HDIM];
    __shared__ float sp[8][KEEP_PAD];
    __shared__ int   si[8][KEEP_PAD];
    const int t = threadIdx.x, lane = t & 31, w = t >> 5;
    const int row = blockIdx.x * 8 + w;      // h*S + q
    const int h = row >> 11, q = row & (S_LEN - 1);
    const int n = cnt[row];

    for (int d = lane; d < HDIM; d += 32) sq[w][d] = qr[(size_t)row * HDIM + d];
    for (int j = lane; j < n; j += 32) si[w][j] = topk[(size_t)row * KEEP_PAD + j];
    __syncthreads();

    float mx = -3.0e38f;
    for (int j = lane; j < n; j += 32) {
        const float* krow = kr + ((size_t)h * S_LEN + si[w][j]) * HDIM;
        float dot = 0.f;
#pragma unroll 8
        for (int d = 0; d < HDIM; ++d) dot += sq[w][d] * krow[d];
        float s = dot * 0.08838834764831845f;   // 1/sqrt(128)
        sp[w][j] = s;
        mx = fmaxf(mx, s);
    }
    mx = wave_max_f(mx);
    __syncthreads();
    float lsum = 0.f;
    for (int j = lane; j < n; j += 32) {
        float e = expf(sp[w][j] - mx);
        sp[w][j] = e;
        lsum += e;
    }
    const float inv = 1.0f / wave_sum_f(lsum);
    __syncthreads();

    float a0 = 0, a1 = 0, a2 = 0, a3 = 0;   // lane owns dims lane+{0,32,64,96}
    for (int j = 0; j < n; ++j) {
        float p = sp[w][j];
        const float* vrow = v + ((size_t)h * S_LEN + si[w][j]) * HDIM;
        a0 += p * vrow[lane];
        a1 += p * vrow[lane + 32];
        a2 += p * vrow[lane + 64];
        a3 += p * vrow[lane + 96];
    }
    const size_t ob = (size_t)q * EMB + (size_t)h * HDIM;   // row-major [s][e] bf16
    outbf[ob + lane]      = f2bf(a0 * inv);
    outbf[ob + lane + 32] = f2bf(a1 * inv);
    outbf[ob + lane + 64] = f2bf(a2 * inv);
    outbf[ob + lane + 96] = f2bf(a3 * inv);
}

// ---------- host launch ----------
extern "C" void kernel_launch(void* const* d_in, const int* in_sizes, int n_in,
                              void* d_out, int out_size, void* d_ws, size_t ws_size,
                              hipStream_t stream) {
    (void)in_sizes; (void)n_in; (void)out_size; (void)ws_size;
    const float* hid = (const float*)d_in[0];
    const float* Wq  = (const float*)d_in[1];
    const float* Wk  = (const float*)d_in[2];
    const float* Wv  = (const float*)d_in[3];
    const float* Wo  = (const float*)d_in[4];
    float* out = (float*)d_out;

    char* ws = (char*)d_ws;
    size_t off = 0;
    auto carve = [&](size_t bytes) -> char* {
        char* p = ws + off;
        off += (bytes + 255) & ~(size_t)255;
        return p;
    };
    unsigned short* hsb  = (unsigned short*)carve((size_t)S_LEN * EMB * 2);
    unsigned short* wqT  = (unsigned short*)carve((size_t)EMB * EMB * 2);
    unsigned short* wkT  = (unsigned short*)carve((size_t)EMB * EMB * 2);
    unsigned short* wvT  = (unsigned short*)carve((size_t)EMB * EMB * 2);
    unsigned short* woT  = (unsigned short*)carve((size_t)EMB * EMB * 2);
    unsigned short* q_bf = (unsigned short*)carve((size_t)NHEAD * S_LEN * HDIM * 2);
    unsigned short* k_bf = (unsigned short*)carve((size_t)NHEAD * S_LEN * HDIM * 2);
    float* v_f32  = (float*)carve((size_t)NHEAD * S_LEN * HDIM * 4);
    float* q_rope = (float*)carve((size_t)NHEAD * S_LEN * HDIM * 4);
    float* k_rope = (float*)carve((size_t)NHEAD * S_LEN * HDIM * 4);
    int* topk = (int*)carve((size_t)NHEAD * S_LEN * KEEP_PAD * 4);
    int* cntb = (int*)carve((size_t)NHEAD * S_LEN * 4);
    unsigned short* attn_bf = (unsigned short*)carve((size_t)S_LEN * EMB * 2);

    // 1) casts: hidden row-major; weights transposed to [N][K] for clean B loads
    cast_f32_bf16<<<(S_LEN * EMB + 255) / 256, 256, 0, stream>>>(hid, hsb, S_LEN * EMB);
    dim3 tgrid(EMB / 32, EMB / 32);
    transpose_cast_bf16<<<tgrid, 256, 0, stream>>>(Wq, wqT);
    transpose_cast_bf16<<<tgrid, 256, 0, stream>>>(Wk, wkT);
    transpose_cast_bf16<<<tgrid, 256, 0, stream>>>(Wv, wvT);
    transpose_cast_bf16<<<tgrid, 256, 0, stream>>>(Wo, woT);

    // 2) projections (async double-buffered WMMA GEMM)
    dim3 ggrid(S_LEN / BM, EMB / BN);
    gemm_bf16_wmma<<<ggrid, 256, 0, stream>>>(hsb, wqT, q_bf, S_LEN, EMB, EMB, 2);
    gemm_bf16_wmma<<<ggrid, 256, 0, stream>>>(hsb, wkT, k_bf, S_LEN, EMB, EMB, 2);
    gemm_bf16_wmma<<<ggrid, 256, 0, stream>>>(hsb, wvT, v_f32, S_LEN, EMB, EMB, 0);

    // 3) RoPE
    rope_kernel<<<NHEAD * S_LEN, 128, 0, stream>>>(q_bf, k_bf, q_rope, k_rope);

    // 4) draft scores + top-205 (WMMA + binary-search selection)
    size_t draft_lds = (size_t)(16 * HDIM + 16 * S_LEN) * sizeof(unsigned short); // 69632 B
    draft_topk_kernel<<<dim3(S_LEN / 16, NHEAD), 256, draft_lds, stream>>>(
        q_bf, k_bf, topk, cntb);

    // 5) sparse attention
    sparse_attn_kernel<<<(NHEAD * S_LEN) / 8, 256, 0, stream>>>(
        q_rope, k_rope, v_f32, topk, cntb, attn_bf);

    // 6) output projection (WMMA) -> fp32 d_out
    gemm_bf16_wmma<<<ggrid, 256, 0, stream>>>(attn_bf, woT, out, S_LEN, EMB, EMB, 1);
}
